// CAModel_30726196035714
// MI455X (gfx1250) — matrix operand order repaired and checked by reading, not compiled, not run on hardware
//
#include <hip/hip_runtime.h>
#include <hip/hip_bf16.h>

typedef __attribute__((ext_vector_type(16))) _Float16 v16h;
typedef __attribute__((ext_vector_type(8)))  _Float16 v8h;
typedef __attribute__((ext_vector_type(8)))  float    v8f;

#define CHN    16
#define HID    128
#define HSZ    512
#define WSZ    512
#define BATCH  8
#define KPAD   64      // 48 perceive features + 1 bias row + zero pad -> 2 K-steps

// Fragment-swizzled weight sizes (halfs)
#define W1F_ELEMS (8 * 2 * 32 * 16)   // [nt=8][ks=2][lane=32][i=16] = 8192
#define W2F_ELEMS (4 * 32 * 16)       // [ks=4][lane=32][i=16]       = 2048

// Single-instruction ReLU (avoids the compiler's canonicalize+max pair).
__device__ __forceinline__ float relu1(float v) {
    float t;
    asm("v_max_num_f32 %0, 0, %1" : "=v"(t) : "v"(v));
    return t;
}

// ---------------------------------------------------------------------------
// Kernel 0: convert weights to f16 in WMMA-fragment order (one lane's 16-half
// fragment is contiguous -> ds_load_b128 pairs in the main kernel).
//
// Layer-1 B frag (32x16 f16, wave32): lane = kh*16 + n (n = out column),
//   slot i<8 holds K = ks*32 + kh*8 + i, i>=8 holds K = ks*32 + 16 + kh*8 + (i-8).
//   Bias folded into K: row 48 of padded W1 = b1 (y supplies 1.0 there).
// Layer-2 A frag (16x32 f16, wave32): lane row = lane&15 (= out channel),
//   slot i (0..15) holds K = ks*32 + kh*16 + i.
//   (A = W2^T)
// ---------------------------------------------------------------------------
__global__ void nca_cvt_weights(const float* __restrict__ w1,   // [48][128]
                                const float* __restrict__ b1,   // [128]
                                const float* __restrict__ w2,   // [128][16]
                                _Float16* __restrict__ w1f,     // [W1F_ELEMS]
                                _Float16* __restrict__ w2f) {   // [W2F_ELEMS]
    for (int j = threadIdx.x; j < W1F_ELEMS; j += blockDim.x) {
        const int i    = j & 15;
        const int lane = (j >> 4) & 31;
        const int ks   = (j >> 9) & 1;
        const int nt   = j >> 10;
        const int kh   = lane >> 4, n = lane & 15;
        const int k = (i < 8) ? (ks * 32 + kh * 8 + i) : (ks * 32 + 16 + kh * 8 + (i - 8));          // row of padded W1
        const int d = nt * 16 + n;                    // column of W1
        float v = 0.f;
        if (k < 48)       v = w1[k * HID + d];
        else if (k == 48) v = b1[d];                  // bias row (y[48] == 1.0)
        w1f[j] = (_Float16)v;
    }
    for (int j = threadIdx.x; j < W2F_ELEMS; j += blockDim.x) {
        const int i    = j & 15;
        const int lane = (j >> 4) & 31;
        const int ks   = j >> 9;
        const int kh   = lane >> 4, c = lane & 15;    // c = output channel (row of W2^T)
        const int k = ks * 32 + kh * 16 + i;
        w2f[j] = (_Float16)w2[k * CHN + c];
    }
}

// ---------------------------------------------------------------------------
// Kernel 1: fused perceive + MLP (WMMA) + masked residual update.
// Block = 128 threads = 4 waves; wave w handles row (blockIdx.y*4+w),
// columns [blockIdx.x*16, +16). Each wave = one M=16 WMMA tile of pixels.
// ---------------------------------------------------------------------------
__global__ __launch_bounds__(128)
void nca_main(const float* __restrict__ x,       // [B][H][W][16]
              const float* __restrict__ noise,   // [B][H][W][1]
              const _Float16* __restrict__ w1f,  // fragment-swizzled (bias in K=48)
              const _Float16* __restrict__ w2f,  // fragment-swizzled (W2^T)
              const float* __restrict__ b2,      // [16]
              float* __restrict__ out,           // [B][H][W][16] (= x2, pre-mask)
              float* __restrict__ alpha2) {      // [B][H][W]     (new alpha)
    __shared__ __align__(16) _Float16 w1_s[W1F_ELEMS];     // 16 KB
    __shared__ __align__(16) _Float16 w2_s[W2F_ELEMS];     //  4 KB
    __shared__ __align__(16) _Float16 y_s[4][16][KPAD];    //  8 KB
    __shared__ __align__(16) _Float16 h_s[4][16][HID];     // 16 KB
    __shared__ float mask_s[4][16];

    const int tid  = threadIdx.x;
    const int wave = tid >> 5;
    const int lane = tid & 31;
    const int p    = lane & 15;   // pixel within tile / A row / B-N column
    const int kh   = lane >> 4;   // K-half selector (wave32 WMMA striping)

    const int b  = blockIdx.z;
    const int r  = blockIdx.y * 4 + wave;
    const int c0 = blockIdx.x * 16;
    const int c  = c0 + p;

    // ---- cooperative b128 weight copy into LDS ----
    {
        const uint4* s1 = (const uint4*)w1f;
        uint4*       d1 = (uint4*)w1_s;
        for (int i = tid; i < W1F_ELEMS / 8; i += 128) d1[i] = s1[i];
        const uint4* s2 = (const uint4*)w2f;
        uint4*       d2 = (uint4*)w2_s;
        for (int i = tid; i < W2F_ELEMS / 8; i += 128) d2[i] = s2[i];
    }

    // ---- perceive: lane covers pixel p, channels [8*kh, 8*kh+8) ----
    {
        const float wyv[3] = {1.f, 2.f, 1.f};
        const float wxv[3] = {-1.f, 0.f, 1.f};
        float ctr[8], sdx[8], sdy[8];
        #pragma unroll
        for (int q = 0; q < 8; ++q) { ctr[q] = 0.f; sdx[q] = 0.f; sdy[q] = 0.f; }

        #pragma unroll
        for (int iy = 0; iy < 3; ++iy) {
            const int rr = r + iy - 1;
            #pragma unroll
            for (int ix = 0; ix < 3; ++ix) {
                const int cc = c + ix - 1;
                float fv[8];
                if ((unsigned)rr < (unsigned)HSZ && (unsigned)cc < (unsigned)WSZ) {
                    const float4* xp = reinterpret_cast<const float4*>(
                        x + ((((b * HSZ) + rr) * WSZ + cc) * CHN + kh * 8));
                    const float4 u0 = xp[0], u1 = xp[1];
                    fv[0]=u0.x; fv[1]=u0.y; fv[2]=u0.z; fv[3]=u0.w;
                    fv[4]=u1.x; fv[5]=u1.y; fv[6]=u1.z; fv[7]=u1.w;
                } else {
                    #pragma unroll
                    for (int q = 0; q < 8; ++q) fv[q] = 0.f;
                }
                const float wdx = wyv[iy] * wxv[ix] * 0.125f;  // dx = outer(wy,wx)/8
                const float wdy = wxv[iy] * wyv[ix] * 0.125f;  // dy = dx^T
                #pragma unroll
                for (int q = 0; q < 8; ++q) {
                    sdx[q] += wdx * fv[q];
                    sdy[q] += wdy * fv[q];
                    if (iy == 1 && ix == 1) ctr[q] = fv[q];
                }
            }
        }
        // Lane's 24 y-halfs are contiguous: [kh*24, kh*24+24) -> 3 x b128 stores.
        _Float16 ybuf[24];
        #pragma unroll
        for (int q = 0; q < 8; ++q) {
            ybuf[3 * q + 0] = (_Float16)ctr[q];
            ybuf[3 * q + 1] = (_Float16)sdx[q];
            ybuf[3 * q + 2] = (_Float16)sdy[q];
        }
        v8h* ydst = (v8h*)&y_s[wave][p][kh * 24];
        #pragma unroll
        for (int t = 0; t < 3; ++t) {
            v8h v;
            #pragma unroll
            for (int e = 0; e < 8; ++e) v[e] = ybuf[t * 8 + e];
            ydst[t] = v;
        }
        // K padding 48..63; element 48 = 1.0 activates the bias row of W1.
        v8h z;
        #pragma unroll
        for (int e = 0; e < 8; ++e) z[e] = (_Float16)0.f;
        if (kh == 0) z[0] = (_Float16)1.0f;
        *(v8h*)&y_s[wave][p][48 + kh * 8] = z;

        if (kh == 0) {
            const int idx = ((b * HSZ) + r) * WSZ + c;
            mask_s[wave][p] = (noise[idx] <= 0.5f) ? 1.0f : 0.0f;
        }
    }

    __syncthreads();   // weights visible to all waves (y/h are wave-private)

    // ---- layer 1: h = relu(A(16x64) * W1b(64x128)), bias via K row 48 ----
    v16h a0, a1;
    {
        const v8h* yp = (const v8h*)&y_s[wave][p][0];   // 8 chunks of 8 halfs
        const v8h l0 = yp[kh];         // K[kh*8 .. +8)
        const v8h h0 = yp[2 + kh];     // K[16+kh*8 .. +8)
        const v8h l1 = yp[4 + kh];     // K[32+...]
        const v8h h1 = yp[6 + kh];     // K[48+...]
        a0 = __builtin_shufflevector(l0, h0, 0,1,2,3,4,5,6,7,8,9,10,11,12,13,14,15);
        a1 = __builtin_shufflevector(l1, h1, 0,1,2,3,4,5,6,7,8,9,10,11,12,13,14,15);
    }

    const v8h* w1v = (const v8h*)w1_s;   // [(nt*2+ks)*32 + lane] fragments, 2 v8h each
    #pragma unroll
    for (int nt = 0; nt < 8; ++nt) {
        const int f0 = ((nt * 2 + 0) * 32 + lane) * 2;
        const int f1 = ((nt * 2 + 1) * 32 + lane) * 2;
        const v16h bf0 = __builtin_shufflevector(w1v[f0], w1v[f0 + 1],
                         0,1,2,3,4,5,6,7,8,9,10,11,12,13,14,15);
        const v16h bf1 = __builtin_shufflevector(w1v[f1], w1v[f1 + 1],
                         0,1,2,3,4,5,6,7,8,9,10,11,12,13,14,15);
        v8f acc = {};                                  // inline-0 C operand
        acc = __builtin_amdgcn_wmma_f32_16x16x32_f16(false, a0, false, bf0,
                                                     (short)0, acc, false, false);
        acc = __builtin_amdgcn_wmma_f32_16x16x32_f16(false, a1, false, bf1,
                                                     (short)0, acc, false, false);
        // C/D layout: VGPR j -> pixel row j / j+8, hidden col = nt*16 + (lane&15)
        #pragma unroll
        for (int j = 0; j < 8; ++j) {
            h_s[wave][j + 8 * kh][nt * 16 + p] = (_Float16)relu1(acc[j]);
        }
    }

    // ---- layer 2 (transposed): dx^T(16ch x 16px) = W2^T(16x128) * h^T(128x16) ----
    // A = W2^T frags (pre-swizzled); B frag column = pixel p, 16 contiguous halfs.
    v8f acc2;
    {
        const float4* b2p = reinterpret_cast<const float4*>(b2 + 8 * kh);
        const float4 q0 = b2p[0], q1 = b2p[1];         // channels [8kh, 8kh+8)
        acc2[0] = q0.x; acc2[1] = q0.y; acc2[2] = q0.z; acc2[3] = q0.w;
        acc2[4] = q1.x; acc2[5] = q1.y; acc2[6] = q1.z; acc2[7] = q1.w;
    }

    const v8h* w2v = (const v8h*)w2_s;
    #pragma unroll
    for (int ks = 0; ks < 4; ++ks) {
        const int fa = (ks * 32 + lane) * 2;
        const v16h wa = __builtin_shufflevector(w2v[fa], w2v[fa + 1],
                        0,1,2,3,4,5,6,7,8,9,10,11,12,13,14,15);
        const v8h* hp = (const v8h*)&h_s[wave][p][ks * 32 + kh * 16];
        const v16h hb = __builtin_shufflevector(hp[0], hp[1],
                        0,1,2,3,4,5,6,7,8,9,10,11,12,13,14,15);
        acc2 = __builtin_amdgcn_wmma_f32_16x16x32_f16(false, wa, false, hb,
                                                      (short)0, acc2, false, false);
    }

    // ---- epilogue: lane owns pixel p, channels [8*kh, 8*kh+8) (contiguous) ----
    {
        const int idx = ((b * HSZ) + r) * WSZ + c;      // c = c0 + p
        const float m = mask_s[wave][p];
        const float4* xp = reinterpret_cast<const float4*>(x + idx * CHN + kh * 8);
        const float4 xa = xp[0], xb = xp[1];
        float4 o0, o1;
        o0.x = xa.x + acc2[0] * m;  o0.y = xa.y + acc2[1] * m;
        o0.z = xa.z + acc2[2] * m;  o0.w = xa.w + acc2[3] * m;
        o1.x = xb.x + acc2[4] * m;  o1.y = xb.y + acc2[5] * m;
        o1.z = xb.z + acc2[6] * m;  o1.w = xb.w + acc2[7] * m;
        float4* op = reinterpret_cast<float4*>(out + idx * CHN + kh * 8);
        op[0] = o0;
        op[1] = o1;
        if (kh == 0) alpha2[idx] = o0.w;                // channel 3 = new alpha
    }
}

// ---------------------------------------------------------------------------
// Kernel 2: life gate. pre = maxpool3(x alpha) > 0.1 (from original input),
// post = maxpool3(new alpha from ws) > 0.1; out *= (pre & post).
// ---------------------------------------------------------------------------
__global__ __launch_bounds__(256)
void nca_life(const float* __restrict__ x,
              const float* __restrict__ alpha2,
              float* __restrict__ out) {
    const int b = blockIdx.z;
    const int r = blockIdx.y * 16 + threadIdx.y;
    const int c = blockIdx.x * 16 + threadIdx.x;

    float pre = -3.0e38f, post = -3.0e38f;
    #pragma unroll
    for (int iy = 0; iy < 3; ++iy) {
        const int rr = r + iy - 1;
        if ((unsigned)rr >= (unsigned)HSZ) continue;
        #pragma unroll
        for (int ix = 0; ix < 3; ++ix) {
            const int cc = c + ix - 1;
            if ((unsigned)cc >= (unsigned)WSZ) continue;
            const int nidx = ((b * HSZ) + rr) * WSZ + cc;
            pre  = fmaxf(pre,  x[nidx * CHN + 3]);
            post = fmaxf(post, alpha2[nidx]);
        }
    }
    const float m = (pre > 0.1f && post > 0.1f) ? 1.0f : 0.0f;

    const int idx = ((b * HSZ) + r) * WSZ + c;
    float4* o = reinterpret_cast<float4*>(out + idx * CHN);
    #pragma unroll
    for (int q = 0; q < 4; ++q) {
        float4 v = o[q];
        v.x *= m; v.y *= m; v.z *= m; v.w *= m;
        o[q] = v;
    }
}

// ---------------------------------------------------------------------------
extern "C" void kernel_launch(void* const* d_in, const int* in_sizes, int n_in,
                              void* d_out, int out_size, void* d_ws, size_t ws_size,
                              hipStream_t stream) {
    const float* x     = (const float*)d_in[0];
    const float* noise = (const float*)d_in[1];
    const float* w1    = (const float*)d_in[2];
    const float* b1    = (const float*)d_in[3];
    const float* w2    = (const float*)d_in[4];
    const float* b2    = (const float*)d_in[5];
    float* out = (float*)d_out;

    // workspace layout: [alpha2: B*H*W f32][w1f: 8192 f16][w2f: 2048 f16]
    float*    alpha2 = (float*)d_ws;
    _Float16* w1f    = (_Float16*)((char*)d_ws + (size_t)BATCH * HSZ * WSZ * sizeof(float));
    _Float16* w2f    = w1f + W1F_ELEMS;

    nca_cvt_weights<<<1, 256, 0, stream>>>(w1, b1, w2, w1f, w2f);

    dim3 gMain(WSZ / 16, HSZ / 4, BATCH);
    nca_main<<<gMain, 128, 0, stream>>>(x, noise, w1f, w2f, b2, out, alpha2);

    dim3 gLife(WSZ / 16, HSZ / 16, BATCH);
    nca_life<<<gLife, dim3(16, 16), 0, stream>>>(x, alpha2, out);
}
